// BackAttention_16398185136493
// MI455X (gfx1250) — matrix operand order repaired and checked by reading, not compiled, not run on hardware
//
#include <hip/hip_runtime.h>
#include <hip/hip_bf16.h>
#include <float.h>

// ---------------------------------------------------------------------------
// BackAttention for MI455X (gfx1250, wave32, WMMA).
// GEMMs: v_wmma_f32_16x16x32_bf16, B-operand tiles double-buffered in LDS via
// async global->LDS copies (ASYNCcnt protocol), A-operand streamed from global
// with global_prefetch_b8. q1/q2, k1/k2, v1/v2 chains composed into single
// 512x512 weights. ab (170MB fp32 ~ L2-resident) is overwritten in place by
// the bf16 masked attention (row pitch 2N) to keep the attention chain in L2.
// Workspace requirement: ~342 MB.
// ---------------------------------------------------------------------------

typedef __bf16 bf16_t;
typedef __attribute__((ext_vector_type(16))) __bf16 v16bf;
typedef __attribute__((ext_vector_type(8)))  __bf16 v8bf;
typedef __attribute__((ext_vector_type(8)))  float  v8f;
typedef __attribute__((ext_vector_type(4)))  int    v4i;

#define BB   8
#define CC   512
#define NN   2304           // H*W = 48*48
#define MTOT (BB*NN)        // 18432

// gemm flags
#define GF_STORE_F32   1
#define GF_STORE_BF16  2
#define GF_STORE_BF16T 4
#define GF_RELU        8
#define GF_BIAS        16

// ---- gfx1250 async global->LDS support (guarded; fallback = sync copy) ----
#if __has_builtin(__builtin_amdgcn_global_load_async_to_lds_b128)
#define HAVE_ASYNC 1
#else
#define HAVE_ASYNC 0
#endif
#if __has_builtin(__builtin_amdgcn_s_wait_asynccnt)
#define HAVE_WAITASYNC 1
#else
#define HAVE_WAITASYNC 0
#endif

// builtin signature (from hipcc diagnostic): param0 = v4i addrspace(1)* (non-const)
typedef __attribute__((address_space(1))) v4i as1_v4i;
typedef __attribute__((address_space(3))) v4i as3_v4i;

__device__ __forceinline__ void async_copy_b128(const bf16_t* g, bf16_t* l)
{
#if HAVE_ASYNC
    __builtin_amdgcn_global_load_async_to_lds_b128(
        (as1_v4i*)g, (as3_v4i*)l, 0, 0);
#else
    *(v8bf*)l = *(const v8bf*)g;     // sync fallback: global load + ds_store
#endif
}

__device__ __forceinline__ void wait_async0()
{
#if HAVE_ASYNC
#if HAVE_WAITASYNC
    __builtin_amdgcn_s_wait_asynccnt(0);
#else
    asm volatile("s_wait_asynccnt 0x0" ::: "memory");
#endif
#endif
}

// ---------------------------------------------------------------------------
// Generic WMMA GEMM: Y[z] = act(alpha * X[z] (MxK) * S[z]^T (Nout x K) + bias)
// Block: 256 threads = 8 waves, tile 32(M) x 256(N); each wave: 16x64.
// S-tile (256 rows x 32 k, bf16) is staged in LDS, double-buffered, filled by
// async global->LDS B128 copies (one S-row per thread, 4 x 16B segments).
// LDS row pitch 80B (64B data + 16B pad) -> conflict-free 16-lane frag reads.
// Fragment lane layouts per CDNA5 ISA 7.12.2:
//   A: lane l: M=l&15, half=l>>4; elems 0..7 -> K=8*half+e; 8..15 -> K=16+8*half+(e-8)
//   B: lane l: N=l&15, half=l>>4; elems 0..7 -> K=8*half+e; 8..15 -> K=16+8*half+(e-8)
//   C/D: vgpr r: M = r + 8*half, N = l&15
// ---------------------------------------------------------------------------
#define TILE_PITCH 80                 // bytes per staged S-row
#define TILE_BYTES (256*TILE_PITCH)   // 20480 per buffer

__global__ __launch_bounds__(256)
void wmma_gemm(const bf16_t* X, long ldx, long bsx,
               const bf16_t* S, long lds_, long bss,
               const float* bias, float alpha, int K, int flags,
               float* Yf, long ldyf, long bsyf,
               bf16_t* Yb, long ldyb, long bsyb,
               bf16_t* Yt, long ldyt, long bsyt, int npb)
{
    __shared__ __attribute__((aligned(32))) unsigned char sb[2 * TILE_BYTES];

    const int z = blockIdx.z;
    const bf16_t* Xb = X + (long)z * bsx;
    const bf16_t* Sb = S + (long)z * bss;

    const int tid  = threadIdx.x;
    const int lane = tid & 31;
    const int wid  = tid >> 5;       // 0..7
    const int wm   = wid >> 2;       // 0..1
    const int wn   = wid & 3;        // 0..3
    const int half = lane >> 4;
    const int l15  = lane & 15;
    const int m0   = blockIdx.y * 32 + wm * 16;
    const int n0w  = wn * 64;                       // wave's col base in tile
    const int n0   = blockIdx.x * 256 + n0w;

    const v8f zacc = {0.f, 0.f, 0.f, 0.f, 0.f, 0.f, 0.f, 0.f};
    v8f acc[4];
#pragma unroll
    for (int j = 0; j < 4; ++j) acc[j] = zacc;

    const bf16_t* xrow = Xb + (long)(m0 + l15) * ldx;
    // staging: thread tid owns S-row (blockIdx.x*256 + tid)
    const bf16_t* srow = Sb + (long)(blockIdx.x * 256 + tid) * lds_;
    bf16_t* lrow0 = (bf16_t*)(sb + tid * TILE_PITCH);
    bf16_t* lrow1 = (bf16_t*)(sb + TILE_BYTES + tid * TILE_PITCH);

    const int nk = K / 32;

    // preload tile 0
#pragma unroll
    for (int seg = 0; seg < 4; ++seg)
        async_copy_b128(srow + seg * 8, lrow0 + seg * 8);
    wait_async0();
    __syncthreads();

    for (int kt = 0; kt < nk; ++kt) {
        const int cur = kt & 1;
        if (kt + 1 < nk) {                       // stage next tile (other buf)
            const bf16_t* src = srow + (kt + 1) * 32;
            bf16_t* dst = cur ? lrow0 : lrow1;
#pragma unroll
            for (int seg = 0; seg < 4; ++seg)
                async_copy_b128(src + seg * 8, dst + seg * 8);
        }

        const int k = kt * 32;
        union { v16bf v; v8bf h[2]; } a;
        a.h[0] = *(const v8bf*)(xrow + k + 8 * half);
        a.h[1] = *(const v8bf*)(xrow + k + 16 + 8 * half);
        if (kt + 1 < nk) __builtin_prefetch(xrow + k + 32, 0, 1);

        const unsigned char* bbase = sb + cur * TILE_BYTES + 16 * half;
#pragma unroll
        for (int j = 0; j < 4; ++j) {
            const unsigned char* bb = bbase + (unsigned)(n0w + j * 16 + l15) * TILE_PITCH;
            union { v16bf v; v8bf h[2]; } bfr;
            bfr.h[0] = *(const v8bf*)(bb);
            bfr.h[1] = *(const v8bf*)(bb + 32);
            acc[j] = __builtin_amdgcn_wmma_f32_16x16x32_bf16(
                false, a.v, false, bfr.v, (short)0, acc[j], false, false);
        }

        wait_async0();        // our async writes to the other buffer done
        __syncthreads();      // all waves done reading/writing this round
    }

#pragma unroll
    for (int j = 0; j < 4; ++j) {
        const int col  = n0 + j * 16 + l15;
        const float bv = (flags & GF_BIAS) ? bias[col] : 0.f;
#pragma unroll
        for (int r = 0; r < 8; ++r) {
            const int row = m0 + r + 8 * half;
            float v = acc[j][r] * alpha + bv;
            if (flags & GF_RELU) v = v > 0.f ? v : 0.f;
            if (flags & GF_STORE_F32)
                Yf[(long)z * bsyf + (long)row * ldyf + col] = v;
            if (flags & GF_STORE_BF16)
                Yb[(long)z * bsyb + (long)row * ldyb + col] = (bf16_t)v;
            if (flags & GF_STORE_BF16T) {
                const int bt = row / npb, ro = row % npb;
                Yt[(long)bt * bsyt + (long)col * ldyt + ro] = (bf16_t)v;
            }
        }
    }
}

// ---------------------------------------------------------------------------
// Per-row softmax + exact stable descending rank (bitonic sort of (val,idx))
// + cubic mask; writes bf16 masked attention IN PLACE inside the ab row
// (row pitch 2*NN bf16 elements == NN fp32). One workgroup per (b,n) row.
// mask = (rank+1)^3 if ab>=0 else 1  (positives always satisfy rank<=apn).
// ---------------------------------------------------------------------------
#define SORTN 4096
__global__ __launch_bounds__(256)
void mask_softmax_kernel(const float* ab, bf16_t* matt)
{
    __shared__ float sv[SORTN];
    __shared__ int   si[SORTN];
    __shared__ float red[256];

    const long row = blockIdx.x;
    const int  tid = threadIdx.x;
    const float* arow = ab + row * (long)NN;
    bf16_t*      mrow = matt + row * (long)(2 * NN);   // aliases arow memory

    float mx = -FLT_MAX;
    for (int i = tid; i < SORTN; i += 256) {
        if (i < NN) { float v = arow[i]; sv[i] = v; si[i] = i; mx = fmaxf(mx, v); }
        else        { sv[i] = -FLT_MAX; si[i] = i; }
    }
    red[tid] = mx; __syncthreads();
    for (int s = 128; s > 0; s >>= 1) {
        if (tid < s) red[tid] = fmaxf(red[tid], red[tid + s]);
        __syncthreads();
    }
    mx = red[0]; __syncthreads();

    float sum = 0.f;
    for (int i = tid; i < NN; i += 256) sum += __expf(sv[i] - mx);
    red[tid] = sum; __syncthreads();
    for (int s = 128; s > 0; s >>= 1) {
        if (tid < s) red[tid] += red[tid + s];
        __syncthreads();
    }
    const float inv = 1.f / red[0];
    __syncthreads();

    // bitonic sort: descending by value, ties broken by ascending index
    for (int k = 2; k <= SORTN; k <<= 1) {
        for (int j = k >> 1; j > 0; j >>= 1) {
            for (int i = tid; i < SORTN; i += 256) {
                const int ixj = i ^ j;
                if (ixj > i) {
                    float va = sv[i], vb = sv[ixj];
                    int   ia = si[i], ib = si[ixj];
                    bool aAfterB = (va < vb) || (va == vb && ia > ib);
                    bool up = ((i & k) == 0);
                    if (up ? aAfterB : !aAfterB) {
                        sv[i] = vb; sv[ixj] = va; si[i] = ib; si[ixj] = ia;
                    }
                }
            }
            __syncthreads();
        }
    }

    for (int p = tid; p < SORTN; p += 256) {
        const int m = si[p];
        if (m < NN) {
            const float v   = sv[p];
            const float att = __expf(v - mx) * inv;
            float mask;
            if (v >= 0.f) { const float rp = (float)(p + 1); mask = rp * rp * rp; }
            else          { mask = 1.f; }
            mrow[m] = (bf16_t)(att * mask);
        }
    }
}

// out_co[n*B+b] = max_c out[b,n,c] + mean_c out[b,n,c]
__global__ __launch_bounds__(256)
void outco_kernel(const float* __restrict__ out, float* __restrict__ co)
{
    __shared__ float rmax[256], rsum[256];
    const long r = blockIdx.x;                 // r = b*NN + n
    const int  b = (int)(r / NN), n = (int)(r % NN);
    const float* orow = out + r * (long)CC;
    const int tid = threadIdx.x;
    const float v0 = orow[tid], v1 = orow[tid + 256];
    rmax[tid] = fmaxf(v0, v1); rsum[tid] = v0 + v1;
    __syncthreads();
    for (int s = 128; s > 0; s >>= 1) {
        if (tid < s) { rmax[tid] = fmaxf(rmax[tid], rmax[tid + s]); rsum[tid] += rsum[tid + s]; }
        __syncthreads();
    }
    if (tid == 0) co[(long)n * BB + b] = rmax[0] + rsum[0] / (float)CC;
}

// softmax over 8 rows of NN (the .view(B,-1) re-interpretation of (N,B) data)
__global__ __launch_bounds__(256)
void softmax_rows_kernel(const float* __restrict__ in, float* __restrict__ o, float scl)
{
    __shared__ float red[256];
    const int b = blockIdx.x, tid = threadIdx.x;
    const float* row = in + (long)b * NN;
    float mx = -FLT_MAX;
    for (int i = tid; i < NN; i += 256) mx = fmaxf(mx, row[i] * scl);
    red[tid] = mx; __syncthreads();
    for (int s = 128; s > 0; s >>= 1) { if (tid < s) red[tid] = fmaxf(red[tid], red[tid + s]); __syncthreads(); }
    mx = red[0]; __syncthreads();
    float sum = 0.f;
    for (int i = tid; i < NN; i += 256) sum += __expf(row[i] * scl - mx);
    red[tid] = sum; __syncthreads();
    for (int s = 128; s > 0; s >>= 1) { if (tid < s) red[tid] += red[tid + s]; __syncthreads(); }
    const float inv = 1.f / red[0];
    for (int i = tid; i < NN; i += 256) o[(long)b * NN + i] = __expf(row[i] * scl - mx) * inv;
}

// pooled[b,c] = (1/N) sum_n xc[b,n,c] * x_co[b,n]
__global__ void pooled_kernel(const float* __restrict__ xc, const float* __restrict__ xco,
                              float* __restrict__ pooled)
{
    const int i = blockIdx.x * blockDim.x + threadIdx.x;
    if (i >= BB * CC) return;
    const int b = i / CC, c = i % CC;
    const float* base = xc + (long)b * NN * CC + c;
    const float* w    = xco + (long)b * NN;
    float s = 0.f;
    for (int n = 0; n < NN; ++n) s += base[(long)n * CC] * w[n];
    pooled[i] = s / (float)NN;
}

// gate[b,o] = sigmoid(pooled[b,:] . atten_w[o,:])
__global__ void gate_kernel(const float* __restrict__ pooled, const float* __restrict__ aw,
                            float* __restrict__ gate)
{
    const int i = blockIdx.x * blockDim.x + threadIdx.x;
    if (i >= BB * CC) return;
    const int b = i / CC, o = i % CC;
    const float* p  = pooled + (long)b * CC;
    const float* wr = aw + (long)o * CC;
    float s = 0.f;
    for (int c = 0; c < CC; ++c) s += p[c] * wr[c];
    gate[i] = 1.f / (1.f + __expf(-s));
}

// out[b,c,n] = xc[b,n,c] * (gate[b,c] + x_co[b,n])
__global__ void final_kernel(const float* __restrict__ xc, const float* __restrict__ gate,
                             const float* __restrict__ xco, float* __restrict__ out)
{
    const long i = blockIdx.x * (long)blockDim.x + threadIdx.x;
    if (i >= (long)BB * CC * NN) return;
    const int n = (int)(i % NN);
    const long t = i / NN;
    const int c = (int)(t % CC), b = (int)(t / CC);
    const float v = xc[((long)b * NN + n) * CC + c];
    out[i] = v * (gate[b * CC + c] + xco[(long)b * NN + n]);
}

// x (B,C,N) fp32 -> xbf (B,N,C) bf16
__global__ void cvt_transpose_kernel(const float* __restrict__ x, bf16_t* __restrict__ xbf)
{
    const long i = blockIdx.x * (long)blockDim.x + threadIdx.x;
    if (i >= (long)BB * CC * NN) return;
    const int n = (int)(i % NN);
    const long t = i / NN;
    const int c = (int)(t % CC), b = (int)(t / CC);
    xbf[((long)b * NN + n) * CC + c] = (bf16_t)x[i];
}

// Wout = bf16(W2 @ W1)   (compose two 1x1 convs)
__global__ void compose_w_kernel(const float* __restrict__ w1, const float* __restrict__ w2,
                                 bf16_t* __restrict__ wout)
{
    const int i = blockIdx.x * blockDim.x + threadIdx.x;
    if (i >= CC * CC) return;
    const int o = i / CC, ii = i % CC;
    const float* r2 = w2 + (long)o * CC;
    float s = 0.f;
    for (int t = 0; t < CC; ++t) s += r2[t] * w1[(long)t * CC + ii];
    wout[i] = (bf16_t)s;
}

// bout = W2 @ b1 + b2
__global__ void compose_b_kernel(const float* __restrict__ w2, const float* __restrict__ b1,
                                 const float* __restrict__ b2, float* __restrict__ bout)
{
    const int o = blockIdx.x * blockDim.x + threadIdx.x;
    if (o >= CC) return;
    const float* r2 = w2 + (long)o * CC;
    float s = b2[o];
    for (int t = 0; t < CC; ++t) s += r2[t] * b1[t];
    bout[o] = s;
}

__global__ void cvt_w_kernel(const float* __restrict__ w, bf16_t* __restrict__ wb)
{
    const int i = blockIdx.x * blockDim.x + threadIdx.x;
    if (i >= CC * CC) return;
    wb[i] = (bf16_t)w[i];
}

// ---------------------------------------------------------------------------
extern "C" void kernel_launch(void* const* d_in, const int* in_sizes, int n_in,
                              void* d_out, int out_size, void* d_ws, size_t ws_size,
                              hipStream_t stream)
{
    const float* x      = (const float*)d_in[0];
    const float* conv_w = (const float*)d_in[1];
    const float* q1_w   = (const float*)d_in[2];
    const float* q2_w   = (const float*)d_in[3];
    const float* k1_w   = (const float*)d_in[4];
    const float* k2_w   = (const float*)d_in[5];
    const float* v1_w   = (const float*)d_in[6];
    const float* v2_w   = (const float*)d_in[7];
    const float* aw     = (const float*)d_in[8];
    const float* conv_b = (const float*)d_in[9];
    const float* q1_b   = (const float*)d_in[10];
    const float* q2_b   = (const float*)d_in[11];
    const float* k1_b   = (const float*)d_in[12];
    const float* k2_b   = (const float*)d_in[13];
    const float* v1_b   = (const float*)d_in[14];
    const float* v2_b   = (const float*)d_in[15];
    float* outp = (float*)d_out;
    (void)in_sizes; (void)n_in; (void)out_size; (void)ws_size;

    const float scale = 0.044194173824159216f;   // 1/sqrt(512)

    // workspace bump allocator (256B aligned slabs); total ~342 MB
    char* p = (char*)d_ws;
    auto alloc = [&](size_t bytes) -> char* {
        char* r = p; p += (bytes + 255) & ~(size_t)255; return r;
    };
    bf16_t* xbf  = (bf16_t*)alloc((size_t)MTOT * CC * 2);
    float*  xc_f = (float*) alloc((size_t)MTOT * CC * 4);
    bf16_t* xc_b = (bf16_t*)alloc((size_t)MTOT * CC * 2);
    bf16_t* qb   = (bf16_t*)alloc((size_t)MTOT * CC * 2);
    bf16_t* kb   = (bf16_t*)alloc((size_t)MTOT * CC * 2);
    bf16_t* vT   = (bf16_t*)alloc((size_t)MTOT * CC * 2);   // (B, C, N) bf16
    float*  abf  = (float*) alloc((size_t)BB * NN * NN * 4); // ab, masked-att aliased
    float*  outf = (float*) alloc((size_t)MTOT * CC * 4);
    float*  co   = (float*) alloc((size_t)MTOT * 4);
    float*  xco  = (float*) alloc((size_t)MTOT * 4);
    float*  pooled = (float*)alloc((size_t)BB * CC * 4);
    float*  gate   = (float*)alloc((size_t)BB * CC * 4);
    bf16_t* cwb = (bf16_t*)alloc((size_t)CC * CC * 2);
    bf16_t* wqb = (bf16_t*)alloc((size_t)CC * CC * 2);
    bf16_t* wkb = (bf16_t*)alloc((size_t)CC * CC * 2);
    bf16_t* wvb = (bf16_t*)alloc((size_t)CC * CC * 2);
    float*  bq  = (float*) alloc((size_t)CC * 4);
    float*  bk  = (float*) alloc((size_t)CC * 4);
    float*  bv  = (float*) alloc((size_t)CC * 4);

    // 1. weight prep: bf16 conv_w, composed q/k/v weights + biases
    cvt_w_kernel<<<(CC * CC + 255) / 256, 256, 0, stream>>>(conv_w, cwb);
    compose_w_kernel<<<(CC * CC + 255) / 256, 256, 0, stream>>>(q1_w, q2_w, wqb);
    compose_w_kernel<<<(CC * CC + 255) / 256, 256, 0, stream>>>(k1_w, k2_w, wkb);
    compose_w_kernel<<<(CC * CC + 255) / 256, 256, 0, stream>>>(v1_w, v2_w, wvb);
    compose_b_kernel<<<2, 256, 0, stream>>>(q2_w, q1_b, q2_b, bq);
    compose_b_kernel<<<2, 256, 0, stream>>>(k2_w, k1_b, k2_b, bk);
    compose_b_kernel<<<2, 256, 0, stream>>>(v2_w, v1_b, v2_b, bv);

    // 2. x (B,C,N) -> bf16 (B,N,C)
    cvt_transpose_kernel<<<(int)(((long)BB * CC * NN + 255) / 256), 256, 0, stream>>>(x, xbf);

    // 3. xc = relu(xf @ conv_w^T + conv_b)  -> fp32 + bf16
    wmma_gemm<<<dim3(CC / 256, MTOT / 32, 1), 256, 0, stream>>>(
        xbf, CC, 0, cwb, CC, 0, conv_b, 1.f, CC,
        GF_STORE_F32 | GF_STORE_BF16 | GF_RELU | GF_BIAS,
        xc_f, CC, 0, xc_b, CC, 0, nullptr, 0, 0, 1);

    // 4. q/k (bf16), v (bf16 transposed to (B,C,N))
    wmma_gemm<<<dim3(CC / 256, MTOT / 32, 1), 256, 0, stream>>>(
        xc_b, CC, 0, wqb, CC, 0, bq, 1.f, CC, GF_STORE_BF16 | GF_BIAS,
        nullptr, 0, 0, qb, CC, 0, nullptr, 0, 0, 1);
    wmma_gemm<<<dim3(CC / 256, MTOT / 32, 1), 256, 0, stream>>>(
        xc_b, CC, 0, wkb, CC, 0, bk, 1.f, CC, GF_STORE_BF16 | GF_BIAS,
        nullptr, 0, 0, kb, CC, 0, nullptr, 0, 0, 1);
    wmma_gemm<<<dim3(CC / 256, MTOT / 32, 1), 256, 0, stream>>>(
        xc_b, CC, 0, wvb, CC, 0, bv, 1.f, CC, GF_STORE_BF16T | GF_BIAS,
        nullptr, 0, 0, nullptr, 0, 0, vT, NN, (long)CC * NN, NN);

    // 5. ab = scale * q @ k^T, per batch (grid.z = B)
    wmma_gemm<<<dim3(NN / 256, NN / 32, BB), 256, 0, stream>>>(
        qb, CC, (long)NN * CC, kb, CC, (long)NN * CC, nullptr, scale, CC,
        GF_STORE_F32, abf, NN, (long)NN * NN, nullptr, 0, 0, nullptr, 0, 0, 1);

    // 6. softmax + rank mask -> bf16 masked attention, in place in ab rows
    mask_softmax_kernel<<<MTOT, 256, 0, stream>>>(abf, (bf16_t*)abf);

    // 7. out = masked_att @ v  (A: bf16 rows pitch 2N inside ab; S = v^T)
    wmma_gemm<<<dim3(CC / 256, NN / 32, BB), 256, 0, stream>>>(
        (const bf16_t*)abf, 2 * NN, (long)NN * 2 * NN,
        vT, NN, (long)CC * NN, nullptr, 1.f, NN,
        GF_STORE_F32, outf, CC, (long)NN * CC, nullptr, 0, 0, nullptr, 0, 0, 1);

    // 8. channel max+mean -> co, softmax over re-viewed rows -> x_co
    outco_kernel<<<MTOT, 256, 0, stream>>>(outf, co);
    softmax_rows_kernel<<<BB, 256, 0, stream>>>(co, xco, scale);

    // 9. squeeze-excite gate and final fused output
    pooled_kernel<<<(BB * CC + 255) / 256, 256, 0, stream>>>(xc_f, xco, pooled);
    gate_kernel<<<(BB * CC + 255) / 256, 256, 0, stream>>>(pooled, aw, gate);
    final_kernel<<<(int)(((long)BB * CC * NN + 255) / 256), 256, 0, stream>>>(
        xc_f, gate, xco, outp);
}
